// Attention_46042049413618
// MI455X (gfx1250) — matrix operand order repaired and checked by reading, not compiled, not run on hardware
//
#include <hip/hip_runtime.h>
#include <hip/hip_bf16.h>

// ---------------------------------------------------------------------------
// Problem constants (from the reference)
// ---------------------------------------------------------------------------
#define DIM_      2048
#define SEQ_      2048
#define BATCH_    2
#define NHEADS_   32
#define NKVH_     8
#define HEADDIM_  64
#define MROWS_    (BATCH_ * SEQ_)   // 4096

typedef __attribute__((ext_vector_type(16))) __bf16 v16bf;
typedef __attribute__((ext_vector_type(8)))  __bf16 v8bf;
typedef __attribute__((ext_vector_type(8)))  float  v8f;

static __device__ __forceinline__ v8f wmma_bf16(v16bf a, v16bf b, v8f c) {
    // D = A(16x32 bf16) * B(32x16 bf16) + C(16x16 f32)
    return __builtin_amdgcn_wmma_f32_16x16x32_bf16(
        /*neg_a=*/false, a, /*neg_b=*/false, b,
        /*c_mod=*/(short)0, c, /*reuse_a=*/false, /*reuse_b=*/false);
}

// ---------------------------------------------------------------------------
// Tensor Data Mover support (guarded; fallback = direct global loads)
// ---------------------------------------------------------------------------
#if defined(__HIP_DEVICE_COMPILE__) &&                                         \
    __has_builtin(__builtin_amdgcn_tensor_load_to_lds) &&                      \
    __has_builtin(__builtin_amdgcn_s_wait_tensorcnt)
#define USE_TDM 1
#else
#define USE_TDM 0
#endif

#if USE_TDM
typedef unsigned int u32x4_ __attribute__((ext_vector_type(4)));
typedef int          i32x8_ __attribute__((ext_vector_type(8)));
typedef int          i32x4_ __attribute__((ext_vector_type(4)));

typedef __attribute__((address_space(3))) void lds_void_t;

// LDS byte offset of a generic pointer to __shared__ memory
static __device__ __forceinline__ unsigned lds_offset_of(const void* p) {
    return (unsigned)(unsigned long long)(lds_void_t*)p;
}

// Issue a 2-D bf16 tile load Global->LDS via the TDM (one TENSORcnt token).
// Dims/strides in elements (data_size = 2 bytes). D# layout per ISA 8.3/8.4.
static __device__ __forceinline__ void tdm_load_2d_bf16(
    unsigned lds_byte, const void* gaddr,
    int tensor_d0, int tensor_d1, int stride0,
    int tile_d0, int tile_d1) {
    unsigned long long ga = (unsigned long long)(uintptr_t)gaddr;
    u32x4_ g0;
    g0[0] = 1u;                                     // count=1, gather off
    g0[1] = lds_byte;                               // lds_addr (bytes)
    g0[2] = (unsigned)(ga & 0xFFFFFFFFu);           // global_addr[31:0]
    g0[3] = (unsigned)((ga >> 32) & 0x1FFFFFFu)     // global_addr[56:32]
            | (2u << 30);                           // type=2 ("image")
    i32x8_ g1;
    g1[0] = (1 << 16);                              // data_size=1 -> 2 bytes
    g1[1] = (tensor_d0 & 0xFFFF) << 16;             // tensor_dim0 lo16 @bit48
    g1[2] = ((tensor_d0 >> 16) & 0xFFFF)            // tensor_dim0 hi16
            | ((tensor_d1 & 0xFFFF) << 16);         // tensor_dim1 lo16
    g1[3] = ((tensor_d1 >> 16) & 0xFFFF)            // tensor_dim1 hi16
            | ((tile_d0 & 0xFFFF) << 16);           // tile_dim0
    g1[4] = (tile_d1 & 0xFFFF);                     // tile_dim1 (tile_dim2=0)
    g1[5] = stride0;                                // tensor_dim0_stride lo32
    g1[6] = 0;                                      // stride0 hi / stride1 lo
    g1[7] = 0;
    i32x4_ gz4 = {0, 0, 0, 0};
#if defined(__clang_major__) && __clang_major__ >= 23
    i32x8_ gz8 = {0, 0, 0, 0, 0, 0, 0, 0};
    __builtin_amdgcn_tensor_load_to_lds(g0, g1, gz4, gz4, gz8, 0);
#else
    __builtin_amdgcn_tensor_load_to_lds(g0, g1, gz4, gz4, 0);
#endif
}
#endif  // USE_TDM

// ---------------------------------------------------------------------------
// f32 -> bf16 conversion
// ---------------------------------------------------------------------------
__global__ void f32_to_bf16_kernel(const float* __restrict__ src,
                                   __bf16* __restrict__ dst, int n) {
    int i = blockIdx.x * blockDim.x + threadIdx.x;
    if (i < n) dst[i] = (__bf16)src[i];
}

// ---------------------------------------------------------------------------
// bf16 GEMM via WMMA:  C[M,N] = A[M,K] * W[N,K]^T   (nn.Linear convention)
// block = 128 threads (4 waves); each wave owns a 32x32 tile of a 64x64 block.
// A/B frags use contiguous k = hi*16 + i (same order in both); C layout per CDNA5 ISA 7.12.2.
// ---------------------------------------------------------------------------
template <typename CT>
__global__ __launch_bounds__(128) void gemm_bf16_wmma(
    const __bf16* __restrict__ A, const __bf16* __restrict__ W,
    CT* __restrict__ C, int M, int N, int K) {
    const int lane = threadIdx.x & 31;
    const int wave = threadIdx.x >> 5;
    const int lo = lane & 15;         // col/row-within-tile lane id
    const int hi = lane >> 4;         // which 16-lane half
    const int r0 = blockIdx.y * 64 + (wave >> 1) * 32;
    const int c0 = blockIdx.x * 64 + (wave & 1) * 32;

    const v8f vz = {0.f, 0.f, 0.f, 0.f, 0.f, 0.f, 0.f, 0.f};
    v8f acc[2][2] = {{vz, vz}, {vz, vz}};

    const __bf16* arow[2] = {A + (size_t)(r0 + lo) * K,
                             A + (size_t)(r0 + 16 + lo) * K};
    const __bf16* wrow[2] = {W + (size_t)(c0 + lo) * K,
                             W + (size_t)(c0 + 16 + lo) * K};

    for (int kk = 0; kk < K; kk += 32) {
        // hint the next tiles into cache (global_prefetch_b8)
        __builtin_prefetch(arow[0] + kk + 128, 0, 1);
        __builtin_prefetch(wrow[0] + kk + 128, 0, 1);

        v16bf a[2], b[2];
#pragma unroll
        for (int mi = 0; mi < 2; ++mi) {
            // A frag: lane holds row (r0+mi*16+lo); elem i -> K=hi*16+i
            // (same contiguous k order as B)
            union { v16bf v; v8bf h[2]; } u;
            u.h[0] = *(const v8bf*)(arow[mi] + kk + hi * 16);
            u.h[1] = *(const v8bf*)(arow[mi] + kk + hi * 16 + 8);
            a[mi] = u.v;
        }
#pragma unroll
        for (int ni = 0; ni < 2; ++ni) {
            // B frag (= W^T chunk): lane holds out-feature (c0+ni*16+lo),
            // elem i -> K = hi*16 + i  => contiguous 32B from W row.
            b[ni] = *(const v16bf*)(wrow[ni] + kk + hi * 16);
        }
#pragma unroll
        for (int mi = 0; mi < 2; ++mi)
#pragma unroll
            for (int ni = 0; ni < 2; ++ni)
                acc[mi][ni] = wmma_bf16(a[mi], b[ni], acc[mi][ni]);
    }

    // C/D layout: vgpr j -> row hi*8+j, lane lo -> col
#pragma unroll
    for (int mi = 0; mi < 2; ++mi)
#pragma unroll
        for (int ni = 0; ni < 2; ++ni)
#pragma unroll
            for (int j = 0; j < 8; ++j) {
                int row = r0 + mi * 16 + hi * 8 + j;
                int col = c0 + ni * 16 + lo;
                C[(size_t)row * N + col] = (CT)acc[mi][ni][j];
            }
}

// ---------------------------------------------------------------------------
// RoPE (in-place, bf16 storage, f32 math). One thread per (b,s,head,i<32) pair.
// ---------------------------------------------------------------------------
__global__ void rope_kernel(__bf16* __restrict__ buf, const int* __restrict__ pos,
                            int nheads, int total) {
    int idx = blockIdx.x * blockDim.x + threadIdx.x;
    if (idx >= total) return;
    int i    = idx & 31;
    int head = (idx >> 5) % nheads;
    int s    = (idx / (32 * nheads)) % SEQ_;
    int b    = idx / (32 * nheads * SEQ_);
    float inv_freq = __powf(10000.0f, -(float)(2 * i) / (float)HEADDIM_);
    float ang = (float)pos[s] * inv_freq;
    float c = __cosf(ang), sn = __sinf(ang);
    __bf16* p = buf + (((size_t)(b * SEQ_ + s)) * nheads + head) * HEADDIM_;
    float x1 = (float)p[i];
    float x2 = (float)p[i + 32];
    p[i]      = (__bf16)(x1 * c - x2 * sn);
    p[i + 32] = (__bf16)(x2 * c + x1 * sn);
}

// ---------------------------------------------------------------------------
// V transpose: Vb[b,s,kvh*64] -> Vt[(b*8+kvh)*64 + d][s]  (so PV B-frags are
// contiguous 32B tile rows for the TDM / vector loads)
// ---------------------------------------------------------------------------
__global__ void transpose_v_kernel(const __bf16* __restrict__ Vb,
                                   __bf16* __restrict__ Vt, int total) {
    int idx = blockIdx.x * blockDim.x + threadIdx.x;
    if (idx >= total) return;
    int d = idx & 511;            // kvh*64 + dd
    int s = (idx >> 9) & (SEQ_ - 1);
    int b = idx >> 20;            // / (2048*512)
    Vt[((size_t)(b * 512 + d)) * SEQ_ + s] =
        Vb[((size_t)(b * SEQ_ + s)) * 512 + d];
}

// ---------------------------------------------------------------------------
// Causal GQA flash attention. One wave (32 threads) per 16-query tile.
// Q A-frags live in registers. Per 32-key chunk: K(32x64) and V^T(64x32)
// tiles are staged into double-buffered LDS by the Tensor Data Mover
// (chunk kc+1 issued while computing chunk kc; TENSORcnt-synchronized),
// then 4 score WMMAs, online softmax (butterfly reductions matching the
// C-tile layout), P re-layout through LDS into an A-frag, 4 PV WMMAs.
// ---------------------------------------------------------------------------
__global__ __launch_bounds__(32) void flash_attn_kernel(
    const __bf16* __restrict__ Qb,   // [B,S,32*64]
    const __bf16* __restrict__ Kb,   // [B,S,8*64]
    const __bf16* __restrict__ Vt,   // [(b*8+kvh)*64+d][S]
    __bf16* __restrict__ Ob) {       // [B,S,32*64]
    __shared__ __attribute__((aligned(32))) __bf16 lds_k[2][32 * 64];
    __shared__ __attribute__((aligned(32))) __bf16 lds_v[2][64 * 32];
    __shared__ __attribute__((aligned(32))) __bf16 lds_p[16 * 32];

    const int lane = threadIdx.x;
    const int lo = lane & 15, hi = lane >> 4;
    const int qt = blockIdx.x;       // 0..127 query tile
    const int h  = blockIdx.y;       // 0..31
    const int b  = blockIdx.z;       // 0..1
    const int hk = h >> 2;           // GQA kv head
    const int qbase = qt * 16;

    // Q A-frags, c = 0/1 covers head-dim K=0..31 / 32..63
    const __bf16* qrow =
        Qb + ((size_t)(b * SEQ_ + qbase + lo)) * (NHEADS_ * HEADDIM_) + h * HEADDIM_;
    v16bf aq[2];
#pragma unroll
    for (int c = 0; c < 2; ++c) {
        union { v16bf v; v8bf hv[2]; } u;
        u.hv[0] = *(const v8bf*)(qrow + c * 32 + hi * 16);
        u.hv[1] = *(const v8bf*)(qrow + c * 32 + hi * 16 + 8);
        aq[c] = u.v;
    }

    float mrow[8], lrow[8];
    const v8f vz = {0.f, 0.f, 0.f, 0.f, 0.f, 0.f, 0.f, 0.f};
    v8f acc[4] = {vz, vz, vz, vz};
#pragma unroll
    for (int j = 0; j < 8; ++j) { mrow[j] = -3.0e38f; lrow[j] = 0.f; }

    const int nkchunks = (qbase + 16 + 31) >> 5;   // ceil((qbase+16)/32)

    const __bf16* kbat = Kb + ((size_t)(b * SEQ_)) * (NKVH_ * HEADDIM_) + hk * HEADDIM_;
    const __bf16* vbat = Vt + ((size_t)((b * NKVH_ + hk) * HEADDIM_)) * SEQ_;

#if USE_TDM
    const unsigned ldsk_off = lds_offset_of(&lds_k[0][0]);
    const unsigned ldsv_off = lds_offset_of(&lds_v[0][0]);
    // TDM: K tile = 32 key-rows x 64 dims (row stride 512 elems);
    //      V^T tile = 64 dim-rows x 32 keys (row stride SEQ_ elems).
    {
        tdm_load_2d_bf16(ldsk_off, kbat, 64, SEQ_, NKVH_ * HEADDIM_, 64, 32);
        tdm_load_2d_bf16(ldsv_off, vbat, SEQ_, 64, SEQ_, 32, 64);
    }
#endif

    for (int kc = 0; kc < nkchunks; ++kc) {
        const int kbase = kc * 32;

#if USE_TDM
        if (kc + 1 < nkchunks) {
            const int kn = kbase + 32;
            const int buf = (kc + 1) & 1;
            tdm_load_2d_bf16(ldsk_off + buf * 4096, kbat + (size_t)kn * (NKVH_ * HEADDIM_),
                             64, SEQ_ - kn, NKVH_ * HEADDIM_, 64, 32);
            tdm_load_2d_bf16(ldsv_off + buf * 4096, vbat + kn,
                             SEQ_ - kn, 64, SEQ_, 32, 64);
            __builtin_amdgcn_s_wait_tensorcnt(2);  // current chunk's 2 ops done
        } else {
            __builtin_amdgcn_s_wait_tensorcnt(0);
        }
        __syncthreads();
        const __bf16* kt = &lds_k[kc & 1][0];
        const __bf16* vt = &lds_v[kc & 1][0];
#endif

        // ---- scores: S[16q x 32k] as two 16x16 WMMA tiles
        v8f s[2] = {vz, vz};
#pragma unroll
        for (int t = 0; t < 2; ++t) {
#pragma unroll
            for (int c = 0; c < 2; ++c) {
#if USE_TDM
                v16bf bk = *(const v16bf*)(kt + (t * 16 + lo) * 64 + c * 32 + hi * 16);
#else
                const __bf16* krow = kbat + (size_t)(kbase + t * 16 + lo) * (NKVH_ * HEADDIM_);
                v16bf bk = *(const v16bf*)(krow + c * 32 + hi * 16);
#endif
                s[t] = wmma_bf16(aq[c], bk, s[t]);
            }
        }

        // ---- scale + causal mask (value select, no divergence)
        const float scale = 0.125f;   // 1/sqrt(64)
#pragma unroll
        for (int t = 0; t < 2; ++t) {
            int key = kbase + t * 16 + lo;
#pragma unroll
            for (int j = 0; j < 8; ++j) {
                int q = qbase + hi * 8 + j;
                float v = s[t][j] * scale;
                s[t][j] = (key <= q) ? v : -3.0e38f;
            }
        }

        // ---- online softmax; row m = hi*8+j spans the 16 lanes of this half
#pragma unroll
        for (int j = 0; j < 8; ++j) {
            float v = fmaxf(s[0][j], s[1][j]);
            v = fmaxf(v, __shfl_xor(v, 1));
            v = fmaxf(v, __shfl_xor(v, 2));
            v = fmaxf(v, __shfl_xor(v, 4));
            v = fmaxf(v, __shfl_xor(v, 8));
            float mn = fmaxf(mrow[j], v);
            float alpha = __expf(mrow[j] - mn);
            mrow[j] = mn;
            float p0 = __expf(s[0][j] - mn);
            float p1 = __expf(s[1][j] - mn);
            s[0][j] = p0; s[1][j] = p1;
            float rs = p0 + p1;
            rs += __shfl_xor(rs, 1);
            rs += __shfl_xor(rs, 2);
            rs += __shfl_xor(rs, 4);
            rs += __shfl_xor(rs, 8);
            lrow[j] = lrow[j] * alpha + rs;
#pragma unroll
            for (int n = 0; n < 4; ++n) acc[n][j] *= alpha;
        }

        // ---- P (C-tile layout) -> LDS -> A-frag layout
#pragma unroll
        for (int t = 0; t < 2; ++t)
#pragma unroll
            for (int j = 0; j < 8; ++j)
                lds_p[(hi * 8 + j) * 32 + t * 16 + lo] = (__bf16)s[t][j];
        __syncthreads();
        union { v16bf v; v8bf hv[2]; } up;
        up.hv[0] = *(const v8bf*)(&lds_p[lo * 32 + hi * 16]);
        up.hv[1] = *(const v8bf*)(&lds_p[lo * 32 + hi * 16 + 8]);
        v16bf ap = up.v;
        __syncthreads();

        // ---- O += P(16x32) * V(32x64): 4 WMMAs
#pragma unroll
        for (int n = 0; n < 4; ++n) {
#if USE_TDM
            v16bf bv = *(const v16bf*)(vt + (n * 16 + lo) * 32 + hi * 16);
#else
            const __bf16* vrow = vbat + (size_t)(n * 16 + lo) * SEQ_ + kbase + hi * 16;
            v16bf bv = *(const v16bf*)vrow;
#endif
            acc[n] = wmma_bf16(ap, bv, acc[n]);
        }
    }

    // ---- normalize + store [B,S,32*64]
#pragma unroll
    for (int n = 0; n < 4; ++n)
#pragma unroll
        for (int j = 0; j < 8; ++j) {
            int row = qbase + hi * 8 + j;
            float o = acc[n][j] / lrow[j];
            Ob[((size_t)(b * SEQ_ + row)) * (NHEADS_ * HEADDIM_) + h * HEADDIM_ +
               n * 16 + lo] = (__bf16)o;
        }
}

// ---------------------------------------------------------------------------
// Host launcher
// ---------------------------------------------------------------------------
extern "C" void kernel_launch(void* const* d_in, const int* in_sizes, int n_in,
                              void* d_out, int out_size, void* d_ws, size_t ws_size,
                              hipStream_t stream) {
    (void)in_sizes; (void)n_in; (void)out_size; (void)ws_size;
    const float* x  = (const float*)d_in[0];
    const int*   pos= (const int*)d_in[1];
    const float* wq = (const float*)d_in[2];
    const float* wk = (const float*)d_in[3];
    const float* wv = (const float*)d_in[4];
    const float* wo = (const float*)d_in[5];
    float* out = (float*)d_out;

    char* ws = (char*)d_ws;
    size_t off = 0;
    auto carve = [&](size_t bytes) -> char* {
        char* p = ws + off;
        off = (off + bytes + 255) & ~(size_t)255;
        return p;
    };
    __bf16* xb  = (__bf16*)carve((size_t)MROWS_ * DIM_ * 2);    // x bf16
    __bf16* wqb = (__bf16*)carve((size_t)DIM_ * DIM_ * 2);
    __bf16* wkb = (__bf16*)carve((size_t)512 * DIM_ * 2);
    __bf16* wvb = (__bf16*)carve((size_t)512 * DIM_ * 2);
    __bf16* wob = (__bf16*)carve((size_t)DIM_ * DIM_ * 2);
    __bf16* Qb  = (__bf16*)carve((size_t)MROWS_ * 2048 * 2);    // [B,S,32*64]
    __bf16* Kb  = (__bf16*)carve((size_t)MROWS_ * 512 * 2);     // [B,S,8*64]
    __bf16* Vb  = (__bf16*)carve((size_t)MROWS_ * 512 * 2);
    __bf16* Vt  = (__bf16*)carve((size_t)MROWS_ * 512 * 2);     // transposed V
    __bf16* AOb = (__bf16*)carve((size_t)MROWS_ * 2048 * 2);    // attn out

    auto cvt = [&](const float* s, __bf16* d, int n) {
        f32_to_bf16_kernel<<<(n + 255) / 256, 256, 0, stream>>>(s, d, n);
    };
    cvt(x,  xb,  MROWS_ * DIM_);
    cvt(wq, wqb, DIM_ * DIM_);
    cvt(wk, wkb, 512 * DIM_);
    cvt(wv, wvb, 512 * DIM_);
    cvt(wo, wob, DIM_ * DIM_);

    // QKV projections (bf16 WMMA, bf16 out)
    gemm_bf16_wmma<__bf16><<<dim3(2048 / 64, MROWS_ / 64), 128, 0, stream>>>(
        xb, wqb, Qb, MROWS_, 2048, DIM_);
    gemm_bf16_wmma<__bf16><<<dim3(512 / 64, MROWS_ / 64), 128, 0, stream>>>(
        xb, wkb, Kb, MROWS_, 512, DIM_);
    gemm_bf16_wmma<__bf16><<<dim3(512 / 64, MROWS_ / 64), 128, 0, stream>>>(
        xb, wvb, Vb, MROWS_, 512, DIM_);

    // RoPE on Q and K
    {
        int totq = BATCH_ * SEQ_ * NHEADS_ * 32;
        rope_kernel<<<(totq + 255) / 256, 256, 0, stream>>>(Qb, pos, NHEADS_, totq);
        int totk = BATCH_ * SEQ_ * NKVH_ * 32;
        rope_kernel<<<(totk + 255) / 256, 256, 0, stream>>>(Kb, pos, NKVH_, totk);
    }

    // V transpose for PV fragment loads
    {
        int tot = BATCH_ * SEQ_ * 512;
        transpose_v_kernel<<<(tot + 255) / 256, 256, 0, stream>>>(Vb, Vt, tot);
    }

    // Flash attention: one wave per 16-query tile
    flash_attn_kernel<<<dim3(SEQ_ / 16, NHEADS_, BATCH_), 32, 0, stream>>>(
        Qb, Kb, Vt, AOb);

    // Output projection -> f32 result
    gemm_bf16_wmma<float><<<dim3(2048 / 64, MROWS_ / 64), 128, 0, stream>>>(
        AOb, wob, out, MROWS_, 2048, DIM_);
}
